// GraphAttention_70205535420827
// MI455X (gfx1250) — matrix operand order repaired and checked by reading, not compiled, not run on hardware
//
#include <hip/hip_runtime.h>

typedef __attribute__((ext_vector_type(2)))  float          v2f;
typedef __attribute__((ext_vector_type(8)))  float          v8f;
typedef __attribute__((ext_vector_type(16))) __bf16         v16bf;
typedef __attribute__((ext_vector_type(16))) unsigned short v16us;
typedef __attribute__((ext_vector_type(4)))  unsigned int   v4u;
typedef __attribute__((ext_vector_type(4)))  int            v4i;
typedef __attribute__((ext_vector_type(8)))  int            v8i;

#define N_NODES 8192
#define FDIM    256
#define TK      32     // j-chunk (K of the bf16 WMMA)
#define NCHUNK  (N_NODES / TK)

#if __has_builtin(__builtin_amdgcn_tensor_load_to_lds) && \
    __has_builtin(__builtin_amdgcn_s_wait_tensorcnt)
  #define GAT_HAS_TDM 1
  #if __has_include(<hip/amd_detail/amd_gfx1250_TDM.h>)
    // amdgpu-toolchain (clang-23/therock): 6-arg form
    #define GAT_TDM_LOAD(g0, g1, z4a, z4b, z8) \
        __builtin_amdgcn_tensor_load_to_lds((g0), (g1), (z4a), (z4b), (z8), 0)
  #else
    // ROCm 7.2 (clang-22): 5-arg form
    #define GAT_TDM_LOAD(g0, g1, z4a, z4b, z8) \
        __builtin_amdgcn_tensor_load_to_lds((g0), (g1), (z4a), (z4b), 0)
  #endif
#else
  #define GAT_HAS_TDM 0
#endif

__device__ __forceinline__ unsigned short f2bf(float x) {
    unsigned u = __builtin_bit_cast(unsigned, x);
    u += 0x7FFFu + ((u >> 16) & 1u);          // round-to-nearest-even
    return (unsigned short)(u >> 16);
}

// ---------------------------------------------------------------------------
// Kernel 1: h = node @ W   (full-precision f32 WMMA 16x16x4)
// ---------------------------------------------------------------------------
__global__ __launch_bounds__(128)
void gat_h_kernel(const float* __restrict__ node, const float* __restrict__ W,
                  float* __restrict__ h)
{
    const int lane = threadIdx.x & 31;
    const int wave = threadIdx.x >> 5;
    const int tile = blockIdx.x * 4 + wave;         // 8192 tiles total
    const int m0   = (tile >> 4) << 4;
    const int n0   = (tile & 15) << 4;
    const int half = lane >> 4;
    const int l15  = lane & 15;

    const float* arow = node + (size_t)(m0 + l15) * FDIM;
    v8f acc = {};
#pragma unroll 4
    for (int k0 = 0; k0 < FDIM; k0 += 4) {
        const int ka = k0 + half * 2;
        v2f a, b;
        a.x = arow[ka];
        a.y = arow[ka + 1];
        b.x = W[(size_t)ka * FDIM + n0 + l15];
        b.y = W[(size_t)(ka + 1) * FDIM + n0 + l15];
        acc = __builtin_amdgcn_wmma_f32_16x16x4_f32(false, a, false, b,
                                                    (short)0, acc, false, false);
    }
#pragma unroll
    for (int k = 0; k < 8; ++k)
        h[(size_t)(m0 + half * 8 + k) * FDIM + n0 + l15] = acc[k];
}

// ---------------------------------------------------------------------------
// Kernel 2: transposed bf16 copy of node:  nT[c][j] = bf16(node[j][c])
// ---------------------------------------------------------------------------
__global__ __launch_bounds__(256)
void gat_bf_kernel(const float* __restrict__ node, unsigned short* __restrict__ nT)
{
    __shared__ unsigned short tile[64][72];
    const int j0 = (blockIdx.x & 127) * 64;         // 128 j-tiles
    const int c0 = (blockIdx.x >> 7) * 64;          // 4  c-tiles
#pragma unroll
    for (int i = 0; i < 16; ++i) {
        const int lin = threadIdx.x + i * 256;      // 0..4095
        const int jj = lin >> 6, cc = lin & 63;
        tile[cc][jj] = f2bf(node[(size_t)(j0 + jj) * FDIM + c0 + cc]);
    }
    __syncthreads();
#pragma unroll
    for (int i = 0; i < 16; ++i) {
        const int lin = threadIdx.x + i * 256;
        const int cc = lin >> 6, jj = lin & 63;
        nT[(size_t)(c0 + cc) * N_NODES + j0 + jj] = tile[cc][jj];
    }
}

// ---------------------------------------------------------------------------
// Kernel 3: s1[i] = h[i,:]@a[0:F], s2[i] = h[i,:]@a[F:2F]   (one wave / row)
// ---------------------------------------------------------------------------
__global__ __launch_bounds__(128)
void gat_s_kernel(const float* __restrict__ h, const float* __restrict__ a,
                  float* __restrict__ s1, float* __restrict__ s2)
{
    const int lane = threadIdx.x & 31;
    const int wave = threadIdx.x >> 5;
    const int row  = blockIdx.x * 4 + wave;
    const float* hr = h + (size_t)row * FDIM;
    float p1 = 0.f, p2 = 0.f;
#pragma unroll
    for (int i = lane; i < FDIM; i += 32) {
        const float hv = hr[i];
        p1 += hv * a[i];
        p2 += hv * a[FDIM + i];
    }
#pragma unroll
    for (int off = 16; off >= 1; off >>= 1) {
        p1 += __shfl_xor(p1, off, 32);
        p2 += __shfl_xor(p2, off, 32);
    }
    if (lane == 0) { s1[row] = p1; s2[row] = p2; }
}

// ---------------------------------------------------------------------------
// Kernel 4: s2max = max_j s2[j]   (single block reduction)
// ---------------------------------------------------------------------------
__global__ __launch_bounds__(256)
void gat_max_kernel(const float* __restrict__ s2, float* __restrict__ s2max)
{
    __shared__ float red[256];
    float m = -3.0e38f;
    for (int i = threadIdx.x; i < N_NODES; i += 256) m = fmaxf(m, s2[i]);
    red[threadIdx.x] = m;
    __syncthreads();
    for (int s = 128; s >= 1; s >>= 1) {
        if ((int)threadIdx.x < s)
            red[threadIdx.x] = fmaxf(red[threadIdx.x], red[threadIdx.x + s]);
        __syncthreads();
    }
    if (threadIdx.x == 0) *s2max = red[0];
}

// ---------------------------------------------------------------------------
// Kernel 5: fused masked-softmax attention + P@node, out = h + C/l
// 8 waves/block: wave pairs share a 16-row tile, splitting the 16 F-column
// tiles 8/8. Double-buffered LDS (2 x 32x256 bf16 chunk, [c][k] rows padded to
// 80 B). Wave 0 keeps one Tensor-DMA in flight: wait chunk i, publish via
// barrier, immediately issue chunk i+1 into the other buffer, so the TDM
// streams the next chunk while all waves run bf16 WMMAs on the current one.
// ---------------------------------------------------------------------------
__global__ __launch_bounds__(256)
void gat_attn_kernel(const unsigned short* __restrict__ nT,
                     const int* __restrict__ adj,
                     const float* __restrict__ h, const float* __restrict__ s1,
                     const float* __restrict__ s2, const float* __restrict__ s2maxp,
                     float* __restrict__ out)
{
    __shared__ __align__(16) unsigned short ldsT[2][FDIM][40];  // 2 x 20 KB

    const int lane    = threadIdx.x & 31;
    const int wave    = threadIdx.x >> 5;            // 0..7
    const int rowTile = blockIdx.x * 4 + (wave >> 1);
    const int colHalf = wave & 1;
    const int m0      = rowTile * 16;
    const int half    = lane >> 4;
    const int l15     = lane & 15;

    const float s2max = *s2maxp;
    const int   mrow  = m0 + l15;
    const float s1v   = s1[mrow];
    float cap = s1v + s2max;
    cap = cap > 0.f ? cap : 0.01f * cap;             // leaky_relu of the bound

    float lsum = 0.f;
    v8f acc[8];
#pragma unroll
    for (int t = 0; t < 8; ++t) acc[t] = (v8f){};

    const int kb = half * 8;                         // lane's first K-run start

#if GAT_HAS_TDM
    // Constant part of the Tensor DMA descriptor (group 1):
    //   data_size=1(2B) | pad_enable | pad_interval=3(16 DW) | pad_amount=3(4 DW)
    //   tensor_dim0=8192, tensor_dim1=256, tile_dim0=32, tile_dim1=256,
    //   tensor_dim0_stride=8192  -> LDS rows of 64 B data + 16 B pad = 80 B
    v8i g1;
    g1[0] = (1 << 16) | (1 << 20) | (3 << 22) | (3 << 25);
    g1[1] = (int)(8192u << 16);                      // tensor_dim0[15:0] @ [63:48]
    g1[2] = (int)(256u << 16);                       // tensor_dim1[15:0] @ [95:80]
    g1[3] = (int)(32u << 16);                        // tile_dim0 @ [127:112]
    g1[4] = 256;                                     // tile_dim1 @ [143:128]
    g1[5] = 8192;                                    // tensor_dim0_stride lo32
    g1[6] = 0;
    g1[7] = 0;
    const v4i z4 = {0, 0, 0, 0};
    const v8i z8 = {0, 0, 0, 0, 0, 0, 0, 0};
    const unsigned lds_base[2] = { (unsigned)(size_t)&ldsT[0][0][0],
                                   (unsigned)(size_t)&ldsT[1][0][0] };

    auto tdm_issue = [&](int j0, unsigned lbase) {
        const unsigned long long ga =
            (unsigned long long)(size_t)nT + (unsigned long long)j0 * 2ull;
        v4u g0;
        g0[0] = 1u;                                  // count=1 (valid descriptor)
        g0[1] = lbase;                               // lds_addr
        g0[2] = (unsigned)ga;                        // global_addr lo
        g0[3] = ((unsigned)(ga >> 32) & 0x1FFFFFFu) | (2u << 30);  // hi | type=2
        GAT_TDM_LOAD(g0, g1, z4, z4, z8);
    };

    if (wave == 0) tdm_issue(0, lds_base[0]);        // prime the pipeline
#endif

    for (int it = 0; it < NCHUNK; ++it) {
        const int j0  = it * TK;
        const int buf = it & 1;
#if GAT_HAS_TDM
        if (wave == 0) __builtin_amdgcn_s_wait_tensorcnt(0);  // chunk it arrived
        __syncthreads();                             // publish; prev compute done
        if (wave == 0 && it + 1 < NCHUNK)
            tdm_issue(j0 + TK, lds_base[buf ^ 1]);   // stream next chunk
#else
        __syncthreads();                             // previous chunk consumed
        for (int idx = threadIdx.x; idx < FDIM * 4; idx += 256) {
            const int c = idx >> 2, g = idx & 3;
            *(uint4*)&ldsT[buf][c][g * 8] =
                *(const uint4*)&nT[(size_t)c * N_NODES + j0 + g * 8];
        }
        __syncthreads();
#endif

        // ---- A operand: masked exp probabilities in 16-bit A layout ----
        const int* arow0 = adj + (size_t)mrow * N_NODES + j0 + kb;
        __builtin_prefetch((const void*)(arow0 + TK), 0, 0);  // next adj chunk
        union { int4 v[4]; int i[16]; } A;
        A.v[0] = *(const int4*)(arow0);
        A.v[1] = *(const int4*)(arow0 + 4);
        A.v[2] = *(const int4*)(arow0 + 16);
        A.v[3] = *(const int4*)(arow0 + 20);
        const float* s2p = s2 + j0 + kb;
        union { float4 v[4]; float f[16]; } S;
        S.v[0] = *(const float4*)(s2p);
        S.v[1] = *(const float4*)(s2p + 4);
        S.v[2] = *(const float4*)(s2p + 16);
        S.v[3] = *(const float4*)(s2p + 20);

        v16us pu;
#pragma unroll
        for (int i = 0; i < 16; ++i) {
            float e = s1v + S.f[i];
            e = e > 0.f ? e : 0.01f * e;
            const float ex = __builtin_amdgcn_exp2f((e - cap) * 1.44269504f);
            const float p  = (A.i[i] > 0) ? ex : 0.f;
            lsum += p;
            pu[i] = f2bf(p);
        }
        const v16bf avec = __builtin_bit_cast(v16bf, pu);

        // ---- B operand from LDS, 8 column tiles, bf16 WMMA ----
        const int kB = half * 8;
#pragma unroll
        for (int t = 0; t < 8; ++t) {
            const int c = (colHalf * 8 + t) * 16 + l15;
            struct U { uint4 a, b; } bu;
            bu.a = *(const uint4*)&ldsT[buf][c][kB];
            bu.b = *(const uint4*)&ldsT[buf][c][kB + 16];
            const v16bf bvec = __builtin_bit_cast(v16bf, bu);
            acc[t] = __builtin_amdgcn_wmma_f32_16x16x32_bf16(false, avec, false, bvec,
                                                             (short)0, acc[t],
                                                             false, false);
        }
    }

    // combine the two half-lane partial row sums: row r lives in lanes r, r+16
    lsum += __shfl_xor(lsum, 16, 32);
    float inv[8];
#pragma unroll
    for (int k = 0; k < 8; ++k) {
        const float rs = __shfl(lsum, k + half * 8, 32);
        inv[k] = rs > 0.f ? 1.0f / rs : 0.f;
    }
#pragma unroll
    for (int t = 0; t < 8; ++t) {
        const int c = (colHalf * 8 + t) * 16 + l15;
#pragma unroll
        for (int k = 0; k < 8; ++k) {
            const size_t o = (size_t)(m0 + half * 8 + k) * FDIM + c;
            out[o] = h[o] + acc[t][k] * inv[k];
        }
    }
}

// ---------------------------------------------------------------------------
extern "C" void kernel_launch(void* const* d_in, const int* in_sizes, int n_in,
                              void* d_out, int out_size, void* d_ws, size_t ws_size,
                              hipStream_t stream) {
    const float* node = (const float*)d_in[0];
    const int*   adj  = (const int*)d_in[1];
    const float* W    = (const float*)d_in[2];
    const float* a    = (const float*)d_in[3];
    float* out = (float*)d_out;

    float* h   = (float*)d_ws;                        // 8 MB
    float* s1  = h + (size_t)N_NODES * FDIM;          // 32 KB
    float* s2  = s1 + N_NODES;                        // 32 KB
    float* s2m = s2 + N_NODES;                        // 64 B (padded)
    unsigned short* nT = (unsigned short*)(s2m + 16); // 4 MB, bf16 node^T

    gat_h_kernel   <<<(N_NODES / 16) * (FDIM / 16) / 4, 128, 0, stream>>>(node, W, h);
    gat_bf_kernel  <<<512,          256, 0, stream>>>(node, nT);
    gat_s_kernel   <<<N_NODES / 4,  128, 0, stream>>>(h, a, s1, s2);
    gat_max_kernel <<<1,            256, 0, stream>>>(s2, s2m);
    gat_attn_kernel<<<N_NODES / 64, 256, 0, stream>>>(nT, adj, h, s1, s2, s2m, out);
}
